// DAGERC_fushion_44727789421060
// MI455X (gfx1250) — compile-verified, matrix operand order checked by reading, not run on hardware
//
#include <hip/hip_runtime.h>
#include <hip/hip_bf16.h>

// ---------------------------------------------------------------------------
// Problem constants (from reference)
// ---------------------------------------------------------------------------
#define Bc     64
#define Nn     100
#define Hh     300
#define EMBc   1024
#define Lc     2
#define NCc    7
#define CDc    768
#define INDIM  (Hh * (Lc + 1) + EMBc)   // 1924
#define Rr     (Bc * Nn)                // 6400 rows

typedef __attribute__((ext_vector_type(16))) _Float16 v16h;
typedef __attribute__((ext_vector_type(8)))  float    v8f;

// Pack one WMMA f16 fragment (ISA 16-bit 16x32 layout) from a row-major f32
// row. p = k0 + hx*8 (multiple of 8); row must be 16B aligned (K % 4 == 0).
// Elements j=0..7  <- row[p .. p+7]   (two b128 loads)
// Elements j=8..15 <- row[p+16 .. p+23]
__device__ __forceinline__ v16h pack16(const float* __restrict__ row, int p)
{
    const float4* q = reinterpret_cast<const float4*>(row + p);
    const float4 x0 = q[0], x1 = q[1], x2 = q[4], x3 = q[5];
    v16h o;
    o[0]  = (_Float16)x0.x; o[1]  = (_Float16)x0.y; o[2]  = (_Float16)x0.z; o[3]  = (_Float16)x0.w;
    o[4]  = (_Float16)x1.x; o[5]  = (_Float16)x1.y; o[6]  = (_Float16)x1.z; o[7]  = (_Float16)x1.w;
    o[8]  = (_Float16)x2.x; o[9]  = (_Float16)x2.y; o[10] = (_Float16)x2.z; o[11] = (_Float16)x2.w;
    o[12] = (_Float16)x3.x; o[13] = (_Float16)x3.y; o[14] = (_Float16)x3.z; o[15] = (_Float16)x3.w;
    return o;
}

// Guarded variant for the single K-tail chunk (runs once per kernel).
__device__ __forceinline__ v16h pack16_guard(const float* __restrict__ row, int kbase, int K)
{
    v16h o;
#pragma unroll
    for (int j = 0; j < 8; ++j) {
        const int k  = kbase + j;
        const int k2 = kbase + 16 + j;
        o[j]     = (_Float16)((k  < K) ? row[k]  : 0.f);
        o[8 + j] = (_Float16)((k2 < K) ? row[k2] : 0.f);
    }
    return o;
}

#define WMMA_F16(a, b, c) \
    __builtin_amdgcn_wmma_f32_16x16x32_f16(false, (a), false, (b), (short)0, (c), false, false)

// ---------------------------------------------------------------------------
// WMMA GEMM:  C(M,N) = act( A(M,K) @ W(N,K)^T + bias(N) ), K % 4 == 0.
// 256 threads = 8 waves; each wave owns a 32x32 tile (2x2 grid of 16x16
// WMMAs: 4 WMMAs per 4 fragment loads -> 1 WMMA / 64B loaded per lane) of a 64x128
// macro-tile. Out-of-range rows/cols use clamped pointers; their garbage only
// reaches store-masked outputs (WMMA output (m,n) mixes only A-row m, B-col n).
// ---------------------------------------------------------------------------
__global__ __launch_bounds__(256)
void k_gemm_bias_act(const float* __restrict__ A, const float* __restrict__ W,
                     const float* __restrict__ bias, float* __restrict__ C,
                     int M, int N, int K, int act)
{
    const int lane = threadIdx.x & 31;
    const int wave = threadIdx.x >> 5;        // 0..7
    const int hx   = lane >> 4;               // K-half select
    const int r    = lane & 15;
    const int m0 = blockIdx.y * 64  + (wave & 1) * 32;
    const int n0 = blockIdx.x * 128 + (wave >> 1) * 32;
    if (m0 >= M || n0 >= N) return;           // wave-uniform

    const int am0 = m0 + r,      am1 = m0 + 16 + r;
    const int wn0 = n0 + r,      wn1 = n0 + 16 + r;
    const float* Arow0 = A + (size_t)(am0 < M ? am0 : M - 1) * K;
    const float* Arow1 = A + (size_t)(am1 < M ? am1 : M - 1) * K;
    const float* Wrow0 = W + (size_t)(wn0 < N ? wn0 : N - 1) * K;
    const float* Wrow1 = W + (size_t)(wn1 < N ? wn1 : N - 1) * K;

    v8f acc00 = {}, acc01 = {}, acc10 = {}, acc11 = {};
    const int Kmain = K & ~31;
    for (int k0 = 0; k0 < Kmain; k0 += 32) {
        __builtin_prefetch(Arow0 + k0 + 256, 0, 3);
        __builtin_prefetch(Arow1 + k0 + 256, 0, 3);
        __builtin_prefetch(Wrow0 + k0 + 256, 0, 3);
        __builtin_prefetch(Wrow1 + k0 + 256, 0, 3);
        const int p = k0 + hx * 8;
        const v16h a0 = pack16(Arow0, p);
        const v16h a1 = pack16(Arow1, p);
        const v16h b0 = pack16(Wrow0, p);
        const v16h b1 = pack16(Wrow1, p);
        acc00 = WMMA_F16(a0, b0, acc00);
        acc01 = WMMA_F16(a0, b1, acc01);
        acc10 = WMMA_F16(a1, b0, acc10);
        acc11 = WMMA_F16(a1, b1, acc11);
    }
    if (Kmain < K) {                           // single guarded tail chunk
        const int kb = Kmain + hx * 8;
        const v16h a0 = pack16_guard(Arow0, kb, K);
        const v16h a1 = pack16_guard(Arow1, kb, K);
        const v16h b0 = pack16_guard(Wrow0, kb, K);
        const v16h b1 = pack16_guard(Wrow1, kb, K);
        acc00 = WMMA_F16(a0, b0, acc00);
        acc01 = WMMA_F16(a0, b1, acc01);
        acc10 = WMMA_F16(a1, b0, acc10);
        acc11 = WMMA_F16(a1, b1, acc11);
    }

#pragma unroll
    for (int v = 0; v < 8; ++v) {
        const int row0 = m0 + v + hx * 8;
        const int row1 = row0 + 16;
        const int col0 = n0 + r;
        const int col1 = col0 + 16;
        const float bb0 = (bias && col0 < N) ? bias[col0] : 0.f;
        const float bb1 = (bias && col1 < N) ? bias[col1] : 0.f;
        float x;
        if (row0 < M && col0 < N) {
            x = acc00[v] + bb0; if (act) x = fmaxf(x, 0.f);
            C[(size_t)row0 * N + col0] = x;
        }
        if (row0 < M && col1 < N) {
            x = acc01[v] + bb1; if (act) x = fmaxf(x, 0.f);
            C[(size_t)row0 * N + col1] = x;
        }
        if (row1 < M && col0 < N) {
            x = acc10[v] + bb0; if (act) x = fmaxf(x, 0.f);
            C[(size_t)row1 * N + col0] = x;
        }
        if (row1 < M && col1 < N) {
            x = acc11[v] + bb1; if (act) x = fmaxf(x, 0.f);
            C[(size_t)row1 * N + col1] = x;
        }
    }
}

// ---------------------------------------------------------------------------
// 3-segment-A WMMA GEMM for the comet projections:
//   C = relu( concat(A0,A1,A2)(M,3*CD) @ W(N,3*CD)^T + bias )
// CD = 768 (multiple of 32): K-chunks never straddle a segment; no tail.
// Same 2x2-per-wave tiling as k_gemm_bias_act.
// ---------------------------------------------------------------------------
__global__ __launch_bounds__(256)
void k_gemm3_bias_relu(const float* __restrict__ A0, const float* __restrict__ A1,
                       const float* __restrict__ A2, const float* __restrict__ W,
                       const float* __restrict__ bias, float* __restrict__ C,
                       int M, int N)
{
    const int lane = threadIdx.x & 31;
    const int wave = threadIdx.x >> 5;
    const int hx   = lane >> 4;
    const int r    = lane & 15;
    const int m0 = blockIdx.y * 64  + (wave & 1) * 32;
    const int n0 = blockIdx.x * 128 + (wave >> 1) * 32;
    if (m0 >= M || n0 >= N) return;

    const int am0 = m0 + r,  am1 = m0 + 16 + r;
    const int wn0 = n0 + r,  wn1 = n0 + 16 + r;
    const size_t aoff0 = (size_t)(am0 < M ? am0 : M - 1) * CDc;
    const size_t aoff1 = (size_t)(am1 < M ? am1 : M - 1) * CDc;
    const float* Wrow0 = W + (size_t)(wn0 < N ? wn0 : N - 1) * (3 * CDc);
    const float* Wrow1 = W + (size_t)(wn1 < N ? wn1 : N - 1) * (3 * CDc);

    v8f acc00 = {}, acc01 = {}, acc10 = {}, acc11 = {};
#pragma unroll 1
    for (int seg = 0; seg < 3; ++seg) {
        const float* Abase = (seg == 0) ? A0 : (seg == 1) ? A1 : A2;
        const float* Arow0 = Abase + aoff0;
        const float* Arow1 = Abase + aoff1;
        const int kw0 = seg * CDc;
        for (int kk = 0; kk < CDc; kk += 32) {
            __builtin_prefetch(Arow0 + kk + 256, 0, 3);
            __builtin_prefetch(Arow1 + kk + 256, 0, 3);
            const int p = kk + hx * 8;
            const v16h a0 = pack16(Arow0, p);
            const v16h a1 = pack16(Arow1, p);
            const v16h b0 = pack16(Wrow0, kw0 + p);
            const v16h b1 = pack16(Wrow1, kw0 + p);
            acc00 = WMMA_F16(a0, b0, acc00);
            acc01 = WMMA_F16(a0, b1, acc01);
            acc10 = WMMA_F16(a1, b0, acc10);
            acc11 = WMMA_F16(a1, b1, acc11);
        }
    }

#pragma unroll
    for (int v = 0; v < 8; ++v) {
        const int row0 = m0 + v + hx * 8;
        const int row1 = row0 + 16;
        const int col0 = n0 + r;
        const int col1 = col0 + 16;
        const float bb0 = (col0 < N) ? bias[col0] : 0.f;
        const float bb1 = (col1 < N) ? bias[col1] : 0.f;
        if (row0 < M && col0 < N) C[(size_t)row0 * N + col0] = fmaxf(acc00[v] + bb0, 0.f);
        if (row0 < M && col1 < N) C[(size_t)row0 * N + col1] = fmaxf(acc01[v] + bb1, 0.f);
        if (row1 < M && col0 < N) C[(size_t)row1 * N + col0] = fmaxf(acc10[v] + bb0, 0.f);
        if (row1 < M && col1 < N) C[(size_t)row1 * N + col1] = fmaxf(acc11[v] + bb1, 0.f);
    }
}

// ---------------------------------------------------------------------------
// Row dot-product: out[r] = dot(X[r,:cols], w[:cols])   (attn q/k scalars)
// cols % 4 == 0 and 16B-aligned rows for all call sites (cols = 300).
// ---------------------------------------------------------------------------
__global__ void k_rowdot(const float* __restrict__ X, const float* __restrict__ w,
                         float* __restrict__ out, int rows, int cols)
{
    const int rI = blockIdx.x * blockDim.x + threadIdx.x;
    if (rI >= rows) return;
    const float4* x4 = reinterpret_cast<const float4*>(X + (size_t)rI * cols);
    const float4* w4 = reinterpret_cast<const float4*>(w);
    float s = 0.f;
    const int c4 = cols >> 2;
    for (int c = 0; c < c4; ++c) {
        const float4 a = x4[c], b = w4[c];
        s += a.x * b.x + a.y * b.y + a.z * b.z + a.w * b.w;
    }
    out[rI] = s;
}

__device__ __forceinline__ float sigm(float x) { return 1.f / (1.f + __expf(-x)); }

// ---------------------------------------------------------------------------
// Sequential DAG scan for one layer. One block per batch element (64 blocks,
// 256 threads). All Hl-dependent GRU gates (giC, ghP) and fixed-key attention
// scalars / value projections (oV0/oV1, q*, k2-4) are precomputed by WMMA
// GEMMs; the scan does only the truly sequential work.
// ---------------------------------------------------------------------------
__global__ __launch_bounds__(256)
void k_dag_scan(const float* __restrict__ Hl,  float* __restrict__ Hb,
                const float* __restrict__ giC, const float* __restrict__ ghP,
                const float* __restrict__ WhhC, const float* __restrict__ bhhC,
                const float* __restrict__ WihP, const float* __restrict__ bihP,
                const float* __restrict__ Wr0,  const float* __restrict__ Wr1,
                const float* __restrict__ xH,   const float* __restrict__ xxH,
                const float* __restrict__ oV0,  const float* __restrict__ oV1,
                float* __restrict__ Vr0,        float* __restrict__ Vr1,
                const float* __restrict__ q1,   const float* __restrict__ q2,
                const float* __restrict__ q3,   const float* __restrict__ q4,
                float* __restrict__ k1c,        const float* __restrict__ k2,
                const float* __restrict__ k3,   const float* __restrict__ k4,
                const float* __restrict__ w1hi,
                const float* __restrict__ b1p,  const float* __restrict__ b2p,
                const float* __restrict__ b3p,  const float* __restrict__ b4p,
                const float* __restrict__ adj,  const float* __restrict__ aadj,
                const float* __restrict__ adj2,
                const float* __restrict__ sm,   const float* __restrict__ sm2)
{
    const int b   = blockIdx.x;
    const int tid = threadIdx.x;
    const size_t bnh = (size_t)b * Nn * Hh;
    const size_t bn3 = (size_t)b * Nn * 3 * Hh;
    const size_t bnn = (size_t)b * Nn * Nn;
    const size_t bn  = (size_t)b * Nn;

    __shared__ float att1[Nn], att2[Nn], att3[Nn], att4[Nn];
    __shared__ float Msh[Hh], hsh[Hh];
    __shared__ float ghc[3 * Hh], gip[3 * Hh];
    __shared__ float red[8];
    __shared__ float pr[256];

    const float B1 = b1p[0], B2 = b2p[0], B3 = b3p[0], B4 = b4p[0];

    for (int i = 0; i < Nn; ++i) {
        if (i == 0) {
            for (int g = tid; g < 3 * Hh; g += 256) { ghc[g] = bhhC[g]; gip[g] = bihP[g]; }
            for (int d = tid; d < Hh; d += 256) Msh[d] = 0.f;
        } else {
            // ---- masked attention logits over prefix j < i (4 heads) ----
            const float Q1 = q1[bn + i], Q2 = q2[bn + i], Q3 = q3[bn + i], Q4 = q4[bn + i];
            for (int j = tid; j < i; j += 256) {
                const size_t ij = bnn + (size_t)i * Nn + j;
                att1[j] = (adj [ij] > 0.f) ? (Q1 + k1c[bn + j] + B1) : -1e30f;
                att2[j] = (aadj[ij] > 0.f) ? (Q2 + k2 [bn + j] + B2) : -1e30f;
                att3[j] = (adj2[ij] > 0.f) ? (Q3 + k3 [bn + j] + B3) : -1e30f;
                att4[j] = (adj [ij] > 0.f) ? (Q4 + k4 [bn + j] + B4) : -1e30f;
            }
            __syncthreads();
            if (tid < 4) {                      // per-head softmax normalizers
                float* a = (tid == 0) ? att1 : (tid == 1) ? att2 : (tid == 2) ? att3 : att4;
                float mx = -1e30f;
                for (int j = 0; j < i; ++j) mx = fmaxf(mx, a[j]);
                float s = 0.f;
                for (int j = 0; j < i; ++j) { const float e = __expf(a[j] - mx); a[j] = e; s += e; }
                red[tid] = (s > 0.f) ? 1.f / s : 0.f;
            }
            __syncthreads();
            {
                const float i1 = red[0], i2 = red[1], i3 = red[2], i4 = red[3];
                for (int j = tid; j < i; j += 256) {
                    att1[j] *= i1; att2[j] *= i2; att3[j] *= i3; att4[j] *= i4;
                }
            }
            __syncthreads();
            // ---- aggregate message M ----
            for (int d = tid; d < Hh; d += 256) {
                float acc = 0.f;
                for (int j = 0; j < i; ++j) {
                    const size_t jd = bnh + (size_t)j * Hh + d;
                    const size_t ij = bnn + (size_t)i * Nn + j;
                    const float s1 = sm[ij], s2 = sm2[ij];
                    acc += att1[j] * (s1 * Vr0[jd] + (1.f - s1) * Vr1[jd]);
                    acc += att2[j] * xH[jd];
                    acc += att3[j] * (s2 * oV0[jd] + (1.f - s2) * oV1[jd]);
                    acc += att4[j] * xxH[jd];
                }
                Msh[d] = acc;
            }
            __syncthreads();
            // ---- M-dependent GRU gates: gh_c = M@WhhC^T+b ; gi_p = M@WihP^T+b
            for (int g = tid; g < 3 * Hh; g += 256) {
                const float4* wc = reinterpret_cast<const float4*>(WhhC + (size_t)g * Hh);
                const float4* wp = reinterpret_cast<const float4*>(WihP + (size_t)g * Hh);
                float ac = 0.f, ap = 0.f;
                for (int d = 0; d < Hh / 4; ++d) {
                    const float4 c4 = wc[d], p4 = wp[d];
                    const float m0v = Msh[4 * d],     m1v = Msh[4 * d + 1];
                    const float m2v = Msh[4 * d + 2], m3v = Msh[4 * d + 3];
                    ac += m0v * c4.x + m1v * c4.y + m2v * c4.z + m3v * c4.w;
                    ap += m0v * p4.x + m1v * p4.y + m2v * p4.z + m3v * p4.w;
                }
                ghc[g] = ac + bhhC[g];
                gip[g] = ap + bihP[g];
            }
        }
        __syncthreads();
        // ---- GRU combine: hnew = gru_c(Q, M) + gru_p(M, Q) ----
        const size_t gi0 = bn3 + (size_t)i * 3 * Hh;
        const size_t hi0 = bnh + (size_t)i * Hh;
        for (int d = tid; d < Hh; d += 256) {
            const float rC = sigm(giC[gi0 + d]           + ghc[d]);
            const float zC = sigm(giC[gi0 + Hh + d]      + ghc[Hh + d]);
            const float nC = tanhf(giC[gi0 + 2 * Hh + d] + rC * ghc[2 * Hh + d]);
            const float hc = (1.f - zC) * nC + zC * Msh[d];
            const float rP = sigm(gip[d]           + ghP[gi0 + d]);
            const float zP = sigm(gip[Hh + d]      + ghP[gi0 + Hh + d]);
            const float nP = tanhf(gip[2 * Hh + d] + rP * ghP[gi0 + 2 * Hh + d]);
            const float hp = (1.f - zP) * nP + zP * Hl[hi0 + d];
            const float h  = hc + hp;
            hsh[d] = h;
            Hb[hi0 + d] = h;
        }
        __syncthreads();
        // ---- node-i postlude: relation values Vr0/Vr1 and key scalar k1 ----
        for (int d = tid; d < Hh; d += 256) {
            const float4* w0p = reinterpret_cast<const float4*>(Wr0 + (size_t)d * Hh);
            const float4* w1p = reinterpret_cast<const float4*>(Wr1 + (size_t)d * Hh);
            float a0 = 0.f, a1 = 0.f;
            for (int e = 0; e < Hh / 4; ++e) {
                const float4 u = w0p[e], v = w1p[e];
                const float h0 = hsh[4 * e],     h1 = hsh[4 * e + 1];
                const float h2 = hsh[4 * e + 2], h3 = hsh[4 * e + 3];
                a0 += h0 * u.x + h1 * u.y + h2 * u.z + h3 * u.w;
                a1 += h0 * v.x + h1 * v.y + h2 * v.z + h3 * v.w;
            }
            Vr0[hi0 + d] = a0;  Vr1[hi0 + d] = a1;
        }
        float p = 0.f;
        for (int e = tid; e < Hh; e += 256) p += hsh[e] * w1hi[e];
        pr[tid] = p;
        __syncthreads();
        for (int s = 128; s > 0; s >>= 1) {
            if (tid < s) pr[tid] += pr[tid + s];
            __syncthreads();
        }
        if (tid == 0) k1c[bn + i] = pr[0];
        __syncthreads();
    }
}

// ---------------------------------------------------------------------------
// Hcat = concat(H0, H1, H2, features) along feature dim -> (B,N,1924)
// ---------------------------------------------------------------------------
__global__ void k_build_hcat(const float* __restrict__ H0, const float* __restrict__ H1,
                             const float* __restrict__ H2, const float* __restrict__ feat,
                             float* __restrict__ out)
{
    const size_t idx = (size_t)blockIdx.x * blockDim.x + threadIdx.x;
    const size_t total = (size_t)Rr * INDIM;
    if (idx >= total) return;
    const size_t row = idx / INDIM;
    const int    c   = (int)(idx % INDIM);
    float v;
    if      (c <     Hh) v = H0[row * Hh + c];
    else if (c < 2 * Hh) v = H1[row * Hh + (c - Hh)];
    else if (c < 3 * Hh) v = H2[row * Hh + (c - 2 * Hh)];
    else                 v = feat[row * EMBc + (c - 3 * Hh)];
    out[idx] = v;
}

// ---------------------------------------------------------------------------
// Final classifier head: logits(R,7) = X(R,300) @ W(7,300)^T + b
// ---------------------------------------------------------------------------
__global__ void k_logits(const float* __restrict__ X, const float* __restrict__ W,
                         const float* __restrict__ bias, float* __restrict__ out, int rows)
{
    const int rI = blockIdx.x * blockDim.x + threadIdx.x;
    if (rI >= rows) return;
    const float4* x4 = reinterpret_cast<const float4*>(X + (size_t)rI * Hh);
    for (int c = 0; c < NCc; ++c) {
        const float4* w4 = reinterpret_cast<const float4*>(W + (size_t)c * Hh);
        float s = bias[c];
        for (int d = 0; d < Hh / 4; ++d) {
            const float4 a = x4[d], w = w4[d];
            s += a.x * w.x + a.y * w.y + a.z * w.z + a.w * w.w;
        }
        out[(size_t)rI * NCc + c] = s;
    }
}

// ---------------------------------------------------------------------------
// Orchestration
// ---------------------------------------------------------------------------
extern "C" void kernel_launch(void* const* d_in, const int* in_sizes, int n_in,
                              void* d_out, int out_size, void* d_ws, size_t ws_size,
                              hipStream_t stream)
{
    (void)in_sizes; (void)n_in; (void)out_size; (void)ws_size;
    auto in = [&](int i) { return (const float*)d_in[i]; };

    // -------- top-level inputs (setup_inputs insertion order) --------
    const float* feat    = in(0);
    const float* xWant   = in(1);
    const float* xEffect = in(2);
    const float* xReact  = in(3);
    const float* oWant   = in(4);
    const float* oEffect = in(5);
    const float* oReact  = in(6);
    const float* xIntent = in(7);
    const float* xAttr   = in(8);
    const float* xNeed   = in(9);
    const float* adj     = in(10);
    const float* aadj    = in(11);
    const float* adj2    = in(12);
    const float* smask   = in(13);
    const float* smask2  = in(14);
    // 15: s_mask_onehot, 16: lengths, 17: uttr  (unused by reference math)

    // -------- params pytree leaves (jax sorted-key flatten), base P=18 -----
    const int P = 18;
    const float* fc1_b = in(P + 0);
    const float* fc1_w = in(P + 1);
    const int LBASE[Lc] = { P + 2, P + 2 + 26 };   // 26 leaves per layer
    const float* mlp_w0 = in(P + 54);
    const float* mlp_b0 = in(P + 55);
    const float* mlp_w1 = in(P + 56);
    const float* mlp_b1 = in(P + 57);
    const float* mlp_w2 = in(P + 58);
    const float* mlp_b2 = in(P + 59);
    const float* ofc_b  = in(P + 60);
    const float* ofc_w  = in(P + 61);
    const float* xfc_b  = in(P + 62);
    const float* xfc_w  = in(P + 63);
    const float* xxfc_b = in(P + 64);
    const float* xxfc_w = in(P + 65);

    // -------- workspace carve-up (floats) --------
    float* ws = (float*)d_ws;
    const size_t RH  = (size_t)Rr * Hh;
    const size_t R3H = RH * 3;
    size_t off = 0;
    auto take = [&](size_t n) { float* p = ws + off; off += n; return p; };
    float* H0buf = take(RH);
    float* H1buf = take(RH);
    float* H2buf = take(RH);
    float* xA  = take(RH);  float* xB  = take(RH);
    float* oA  = take(RH);  float* oB  = take(RH);
    float* xxA = take(RH);  float* xxB = take(RH);
    float* giC = take(R3H);
    float* ghP = take(R3H);
    float* oV0 = take(RH);  float* oV1 = take(RH);
    float* Vr0 = take(RH);  float* Vr1 = take(RH);
    float* q1 = take(Rr); float* q2 = take(Rr); float* q3 = take(Rr); float* q4 = take(Rr);
    float* k1 = take(Rr); float* k2 = take(Rr); float* k3 = take(Rr); float* k4 = take(Rr);
    float* x1 = giC;   // MLP temporaries reuse the (now-free) gate buffers
    float* x2 = ghP;

    const dim3 blk(256);
    auto gg = [](int M, int N) { return dim3((unsigned)((N + 127) / 128), (unsigned)((M + 63) / 64)); };
    const dim3 rdg((Rr + 255) / 256);

    // -------- front-end projections (WMMA GEMMs) --------
    k_gemm_bias_act<<<gg(Rr, Hh), blk, 0, stream>>>(feat, fc1_w, fc1_b, H0buf, Rr, Hh, EMBc, 1);
    k_gemm3_bias_relu<<<gg(Rr, Hh), blk, 0, stream>>>(xWant,   xEffect, xReact, xfc_w,  xfc_b,  xA,  Rr, Hh);
    k_gemm3_bias_relu<<<gg(Rr, Hh), blk, 0, stream>>>(oWant,   oEffect, oReact, ofc_w,  ofc_b,  oA,  Rr, Hh);
    k_gemm3_bias_relu<<<gg(Rr, Hh), blk, 0, stream>>>(xIntent, xAttr,   xNeed,  xxfc_w, xxfc_b, xxA, Rr, Hh);

    // -------- DAG layers --------
    float* Hs[3] = { H0buf, H1buf, H2buf };
    float* xcur = xA;   float* xnxt = xB;
    float* ocur = oA;   float* onxt = oB;
    float* xxcur = xxA; float* xxnxt = xxB;

    for (int l = 0; l < Lc; ++l) {
        const int lb = LBASE[l];
        const float* Wr0w = in(lb + 0);
        const float* Wr1w = in(lb + 1);
        const float* b1p  = in(lb + 2);
        const float* b2p  = in(lb + 3);
        const float* b3p  = in(lb + 4);
        const float* b4p  = in(lb + 5);
        const float* oWr0 = in(lb + 6);
        const float* oWr1 = in(lb + 7);
        const float* w1   = in(lb + 8);
        const float* w2   = in(lb + 9);
        const float* w3   = in(lb + 10);
        const float* w4   = in(lb + 11);
        const float* WhhC = in(lb + 12);
        const float* WihC = in(lb + 13);
        const float* bhhC = in(lb + 14);
        const float* bihC = in(lb + 15);
        const float* WhhP = in(lb + 16);
        const float* WihP = in(lb + 17);
        const float* bhhP = in(lb + 18);
        const float* bihP = in(lb + 19);
        const float* ofcs_b  = in(lb + 20);
        const float* ofcs_w  = in(lb + 21);
        const float* xfcs_b  = in(lb + 22);
        const float* xfcs_w  = in(lb + 23);
        const float* xxfcs_b = in(lb + 24);
        const float* xxfcs_w = in(lb + 25);

        const float* Hin  = Hs[l];
        float*       Hout = Hs[l + 1];

        // per-layer feature refresh (300x300 WMMA GEMMs)
        k_gemm_bias_act<<<gg(Rr, Hh), blk, 0, stream>>>(xcur,  xfcs_w,  xfcs_b,  xnxt,  Rr, Hh, Hh, 1);
        k_gemm_bias_act<<<gg(Rr, Hh), blk, 0, stream>>>(ocur,  ofcs_w,  ofcs_b,  onxt,  Rr, Hh, Hh, 1);
        k_gemm_bias_act<<<gg(Rr, Hh), blk, 0, stream>>>(xxcur, xxfcs_w, xxfcs_b, xxnxt, Rr, Hh, Hh, 1);

        // hoisted GRU gates (depend only on layer input Hl)
        k_gemm_bias_act<<<gg(Rr, 3 * Hh), blk, 0, stream>>>(Hin, WihC, bihC, giC, Rr, 3 * Hh, Hh, 0);
        k_gemm_bias_act<<<gg(Rr, 3 * Hh), blk, 0, stream>>>(Hin, WhhP, bhhP, ghP, Rr, 3 * Hh, Hh, 0);

        // hoisted relation-aware values for the fixed oH keys
        k_gemm_bias_act<<<gg(Rr, Hh), blk, 0, stream>>>(onxt, oWr0, nullptr, oV0, Rr, Hh, Hh, 0);
        k_gemm_bias_act<<<gg(Rr, Hh), blk, 0, stream>>>(onxt, oWr1, nullptr, oV1, Rr, Hh, Hh, 0);

        // hoisted attention query/key scalars
        k_rowdot<<<rdg, blk, 0, stream>>>(Hin,   w1,      q1, Rr, Hh);
        k_rowdot<<<rdg, blk, 0, stream>>>(xnxt,  w2,      q2, Rr, Hh);
        k_rowdot<<<rdg, blk, 0, stream>>>(xnxt,  w2 + Hh, k2, Rr, Hh);
        k_rowdot<<<rdg, blk, 0, stream>>>(xnxt,  w3,      q3, Rr, Hh);
        k_rowdot<<<rdg, blk, 0, stream>>>(onxt,  w3 + Hh, k3, Rr, Hh);
        k_rowdot<<<rdg, blk, 0, stream>>>(xxnxt, w4,      q4, Rr, Hh);
        k_rowdot<<<rdg, blk, 0, stream>>>(xxnxt, w4 + Hh, k4, Rr, Hh);

        // sequential scan (one block per batch element)
        k_dag_scan<<<Bc, blk, 0, stream>>>(Hin, Hout, giC, ghP, WhhC, bhhC, WihP, bihP,
                                           Wr0w, Wr1w, xnxt, xxnxt, oV0, oV1, Vr0, Vr1,
                                           q1, q2, q3, q4, k1, k2, k3, k4, w1 + Hh,
                                           b1p, b2p, b3p, b4p, adj, aadj, adj2, smask, smask2);

        { float* t = xcur;  xcur = xnxt;   xnxt = t; }
        { float* t = ocur;  ocur = onxt;   onxt = t; }
        { float* t = xxcur; xxcur = xxnxt; xxnxt = t; }
    }

    // -------- outputs: d_out = [logits (R*7)] ++ [Hcat (R*1924)] --------
    float* logits = (float*)d_out;
    float* Hcat   = logits + (size_t)Rr * NCc;

    const size_t hcTotal = (size_t)Rr * INDIM;
    k_build_hcat<<<(unsigned)((hcTotal + 255) / 256), blk, 0, stream>>>(H0buf, H1buf, H2buf, feat, Hcat);

    // MLP head: 1924 -> 300 -> 300 -> 7
    k_gemm_bias_act<<<gg(Rr, Hh), blk, 0, stream>>>(Hcat, mlp_w0, mlp_b0, x1, Rr, Hh, INDIM, 1);
    k_gemm_bias_act<<<gg(Rr, Hh), blk, 0, stream>>>(x1,   mlp_w1, mlp_b1, x2, Rr, Hh, Hh,    1);
    k_logits<<<rdg, blk, 0, stream>>>(x2, mlp_w2, mlp_b2, logits, Rr);
}
